// HeteroGCN_3410204033263
// MI455X (gfx1250) — matrix operand order repaired and checked by reading, not compiled, not run on hardware
//
#include <hip/hip_runtime.h>
#include <math.h>

// ---------------------------------------------------------------------------
// HeteroGCN on gfx1250 (MI455X): bf16 WMMA GEMMs + atomic scatter-mean.
// ---------------------------------------------------------------------------

#define NTYPE   9
#define NNODES  50000
#define NEDGES  800000
#define HDIM    128
#define NREL    15

typedef __attribute__((ext_vector_type(16))) __bf16 v16bf;
typedef __attribute__((ext_vector_type(8)))  float  v8f;

// REL = [(0,1),(1,0),(0,2),(2,0),(1,2),(0,3),(3,0),(3,7),(0,4),(4,0),(0,8),(1,4),(0,6),(0,5),(5,0)]
__constant__ int c_rel_dst[NREL] = {1,0,2,0,2,3,0,7,4,0,8,4,6,5,0};

// ---------------------------------------------------------------------------
__global__ void zero_kernel(float* __restrict__ p, long long n) {
    long long stride = (long long)gridDim.x * blockDim.x;
    for (long long i = (long long)blockIdx.x * blockDim.x + threadIdx.x; i < n; i += stride)
        p[i] = 0.0f;
}

// Degrees for all 15 relations at once (edges layout: [15][2][E] int32).
__global__ void degree_kernel(const int* __restrict__ edges, float* __restrict__ deg) {
    long long total  = (long long)NREL * NEDGES;
    long long stride = (long long)gridDim.x * blockDim.x;
    for (long long i = (long long)blockIdx.x * blockDim.x + threadIdx.x; i < total; i += stride) {
        int r = (int)(i / NEDGES);
        int e = (int)(i % NEDGES);
        int d = edges[(long long)r * 2 * NEDGES + NEDGES + e];
        atomicAdd(&deg[r * NNODES + d], 1.0f);
    }
}

__global__ void invdeg_kernel(float* __restrict__ deg) {
    int i = blockIdx.x * blockDim.x + threadIdx.x;
    if (i < NREL * NNODES) deg[i] = 1.0f / fmaxf(deg[i], 1.0f);
}

// WrSum[t] = sum_{r: dst(r)==t} Wr[r]  (K x 128);  bSum[t] = sum b[r] (128)
__global__ void wrsum_kernel(const float* __restrict__ Wr, const float* __restrict__ b,
                             float* __restrict__ wrsum, float* __restrict__ bsum, int K) {
    int i = blockIdx.x * blockDim.x + threadIdx.x;
    int per = K * HDIM;
    if (i < NTYPE * per) {
        int t = i / per, kn = i % per;
        float s = 0.0f;
        for (int r = 0; r < NREL; ++r)
            if (c_rel_dst[r] == t) s += Wr[(long long)r * per + kn];
        wrsum[i] = s;
    }
    if (i < NTYPE * HDIM) {
        int t = i / HDIM, nn = i % HDIM;
        float s = 0.0f;
        for (int r = 0; r < NREL; ++r)
            if (c_rel_dst[r] == t) s += b[r * HDIM + nn];
        bsum[i] = s;
    }
}

// scatter-add: MSG[dst[e]][c] += X[src[e]][c]   (D = 64 or 128, power of two)
__global__ void scatter_kernel(const float* __restrict__ X, const int* __restrict__ src,
                               const int* __restrict__ dst, float* __restrict__ msg, int logD) {
    long long total  = (long long)NEDGES << logD;
    long long stride = (long long)gridDim.x * blockDim.x;
    int dm = (1 << logD) - 1;
    for (long long i = (long long)blockIdx.x * blockDim.x + threadIdx.x; i < total; i += stride) {
        int e = (int)(i >> logD);
        int c = (int)(i & dm);
        int s = src[e], d = dst[e];
        atomicAdd(&msg[((long long)d << logD) + c], X[((long long)s << logD) + c]);
    }
}

// ---------------------------------------------------------------------------
// WMMA GEMM:  C[M x 128] (+)= (A .* invdeg_row)[M x K] @ W[K x 128]  (+ bias)
// wave32; wave w of 8 owns column tile w; 5 row-tiles per block (625 blocks).
// A converted f32 -> bf16 on load; accumulation in f32 via v_wmma.
// All hot-loop indexing is 32-bit and C uses a single base pointer so the
// 8 row accesses (constant 512B stride) fold into instruction offsets.
// ---------------------------------------------------------------------------
template <int K, bool INIT>
__global__ __launch_bounds__(256) void gemm_wmma_kernel(
    const float* __restrict__ A, const float* __restrict__ invdeg,
    const float* __restrict__ W, const float* __restrict__ bias,
    float* __restrict__ C, int m16, int rt_per_block) {

    const int lane = threadIdx.x & 31;
    const int ct   = threadIdx.x >> 5;   // column tile 0..7
    const int li   = lane & 15;
    const int h    = lane >> 4;          // half of wave
    const int n    = ct * 16 + li;       // output column
    constexpr int KS = K / 32;

    // B fragments (contiguous k order, same as A): elem j -> K = ks*32 + 16*h + j
    v16bf fb[KS];
#pragma unroll
    for (int ks = 0; ks < KS; ++ks) {
        const int k0 = ks * 32 + 16 * h;
#pragma unroll
        for (int j = 0; j < 16; ++j)
            fb[ks][j] = (__bf16)W[(k0 + j) * HDIM + n];
    }

    const int rt0 = blockIdx.x * rt_per_block;
    for (int rt = rt0; rt < rt0 + rt_per_block && rt < m16; ++rt) {
        const int   rowA  = rt * 16 + li;
        const float scale = invdeg ? invdeg[rowA] : 1.0f;

        // single base pointer for this lane's 8 C elements (rows stride 128 f32)
        float* __restrict__ cp = C + ((rt * 16 + 8 * h) * HDIM + n);

        v8f c;
        if (INIT) {
            const float bv = bias ? bias[n] : 0.0f;
#pragma unroll
            for (int v = 0; v < 8; ++v) c[v] = bv;
        } else {
#pragma unroll
            for (int v = 0; v < 8; ++v) c[v] = cp[v * HDIM];
        }

        const float* __restrict__ ar = A + rowA * K;
#pragma unroll
        for (int ks = 0; ks < KS; ++ks) {
            // A fragment (16-bit A 16x32): elem j -> k = 16h + j
            const float* __restrict__ ap = ar + ks * 32 + 16 * h;
            v16bf a;
#pragma unroll
            for (int j = 0; j < 8; ++j) a[j]     = (__bf16)(ap[j]      * scale);
#pragma unroll
            for (int j = 0; j < 8; ++j) a[8 + j] = (__bf16)(ap[8 + j] * scale);
            c = __builtin_amdgcn_wmma_f32_16x16x32_bf16(false, a, false, fb[ks],
                                                        (short)0, c, false, false);
        }

#pragma unroll
        for (int v = 0; v < 8; ++v) cp[v * HDIM] = c[v];
    }
}

// ---------------------------------------------------------------------------
// leaky_relu(x/cnt) -> layernorm -> *g + be     (one wave per node)
// ---------------------------------------------------------------------------
__global__ __launch_bounds__(256) void act_norm_kernel(
    const float* __restrict__ X, float* __restrict__ Y,
    const float* __restrict__ g, const float* __restrict__ be,
    float inv_cnt, int n_nodes) {
    int wave = (int)((blockIdx.x * (long long)blockDim.x + threadIdx.x) >> 5);
    int lane = threadIdx.x & 31;
    if (wave >= n_nodes) return;
    const float* xp = X + (long long)wave * HDIM;
    float v[4], s = 0.0f, sq = 0.0f;
#pragma unroll
    for (int j = 0; j < 4; ++j) {
        float t = xp[lane + 32 * j] * inv_cnt;
        t = (t >= 0.0f) ? t : 0.01f * t;
        v[j] = t; s += t; sq += t * t;
    }
#pragma unroll
    for (int m = 16; m >= 1; m >>= 1) {
        s  += __shfl_xor(s,  m, 32);
        sq += __shfl_xor(sq, m, 32);
    }
    float mean = s * (1.0f / HDIM);
    float var  = sq * (1.0f / HDIM) - mean * mean;
    float inv  = rsqrtf(var + 1e-5f);
    float* yp = Y + (long long)wave * HDIM;
#pragma unroll
    for (int j = 0; j < 4; ++j) {
        int idx = lane + 32 * j;
        yp[idx] = (v[j] - mean) * inv * g[idx] + be[idx];
    }
}

// sigmoid(concat(xs) @ Wfc + bfc)   (one wave per node, dot of length 9*128)
__global__ __launch_bounds__(256) void final_kernel(
    const float* __restrict__ X, const float* __restrict__ Wfc,
    const float* __restrict__ bfc, float* __restrict__ out, int n_nodes) {
    int wave = (int)((blockIdx.x * (long long)blockDim.x + threadIdx.x) >> 5);
    int lane = threadIdx.x & 31;
    if (wave >= n_nodes) return;
    float acc = 0.0f;
    for (int t = 0; t < NTYPE; ++t) {
        const float* xp = X + ((long long)t * n_nodes + wave) * HDIM;
        const float* wp = Wfc + t * HDIM;
#pragma unroll
        for (int j = 0; j < 4; ++j) {
            int hdx = lane + 32 * j;
            acc += xp[hdx] * wp[hdx];
        }
    }
#pragma unroll
    for (int m = 16; m >= 1; m >>= 1) acc += __shfl_xor(acc, m, 32);
    if (lane == 0) out[wave] = 1.0f / (1.0f + expf(-(acc + bfc[0])));
}

// ---------------------------------------------------------------------------
static void launch_gemm(int K, bool init, const float* A, const float* inv,
                        const float* W, const float* bias, float* C, hipStream_t s) {
    const int m16 = NNODES / 16;   // 3125 row tiles, exact
    const int rt  = 5;             // 625 blocks * 5 = 3125
    dim3 grid(m16 / rt), block(256);
    if (K == 64) {
        if (init) gemm_wmma_kernel<64,  true ><<<grid, block, 0, s>>>(A, inv, W, bias, C, m16, rt);
        else      gemm_wmma_kernel<64,  false><<<grid, block, 0, s>>>(A, inv, W, bias, C, m16, rt);
    } else {
        if (init) gemm_wmma_kernel<128, true ><<<grid, block, 0, s>>>(A, inv, W, bias, C, m16, rt);
        else      gemm_wmma_kernel<128, false><<<grid, block, 0, s>>>(A, inv, W, bias, C, m16, rt);
    }
}

extern "C" void kernel_launch(void* const* d_in, const int* in_sizes, int n_in,
                              void* d_out, int out_size, void* d_ws, size_t ws_size,
                              hipStream_t stream) {
    const float* x     = (const float*)d_in[0];   // [9][N][64]
    const int*   edges = (const int*)  d_in[1];   // [15][2][E]
    const float* Wl[3] = {(const float*)d_in[2], (const float*)d_in[5], (const float*)d_in[8]};
    const float* Wr[3] = {(const float*)d_in[3], (const float*)d_in[6], (const float*)d_in[9]};
    const float* bb[3] = {(const float*)d_in[4], (const float*)d_in[7], (const float*)d_in[10]};
    const float* g[3]  = {(const float*)d_in[11], (const float*)d_in[13], (const float*)d_in[15]};
    const float* be[3] = {(const float*)d_in[12], (const float*)d_in[14], (const float*)d_in[16]};
    const float* Wfc   = (const float*)d_in[17];
    const float* bfc   = (const float*)d_in[18];
    float*       out   = (float*)d_out;

    // workspace layout (floats)
    float* ws     = (float*)d_ws;
    float* bufA   = ws;                                          // [9][N][128]
    float* bufB   = bufA  + (long long)NTYPE * NNODES * HDIM;    // [9][N][128]
    float* msg    = bufB  + (long long)NTYPE * NNODES * HDIM;    // [N][128]
    float* invdeg = msg   + (long long)NNODES * HDIM;            // [15][N]
    float* wrsum  = invdeg + (long long)NREL * NNODES;           // [9][128][128]
    float* bsum   = wrsum + (long long)NTYPE * HDIM * HDIM;      // [9][128]

    static const int REL_S[NREL] = {0,1,0,2,1,0,3,3,0,4,0,1,0,0,5};
    static const int REL_D[NREL] = {1,0,2,0,2,3,0,7,4,0,8,4,6,5,0};
    static const int CNT[NTYPE]  = {5,1,2,1,2,1,1,1,1};

    // --- inverse degrees (shared across all 3 layers) ---
    zero_kernel<<<2048, 256, 0, stream>>>(invdeg, (long long)NREL * NNODES);
    degree_kernel<<<8192, 256, 0, stream>>>(edges, invdeg);
    invdeg_kernel<<<(NREL * NNODES + 255) / 256, 256, 0, stream>>>(invdeg);

    for (int l = 0; l < 3; ++l) {
        const int K = (l == 0) ? 64 : HDIM;
        const int logD = (l == 0) ? 6 : 7;
        const float* Xin = (l == 0) ? x : bufA;

        // combined self-term weights per dst type
        wrsum_kernel<<<(NTYPE * K * HDIM + 255) / 256, 256, 0, stream>>>(
            Wr[l], bb[l], wrsum, bsum, K);

        // ACC[t] = X[t] @ WrSum[t] + bSum[t]
        for (int t = 0; t < NTYPE; ++t)
            launch_gemm(K, true,
                        Xin + (long long)t * NNODES * K, nullptr,
                        wrsum + (long long)t * K * HDIM, bsum + t * HDIM,
                        bufB + (long long)t * NNODES * HDIM, stream);

        // per relation: scatter-sum then ACC[d] += (msg * invdeg) @ Wl[r]
        for (int r = 0; r < NREL; ++r) {
            zero_kernel<<<4096, 256, 0, stream>>>(msg, (long long)NNODES * K);
            scatter_kernel<<<8192, 256, 0, stream>>>(
                Xin + (long long)REL_S[r] * NNODES * K,
                edges + (long long)r * 2 * NEDGES,
                edges + (long long)r * 2 * NEDGES + NEDGES,
                msg, logD);
            launch_gemm(K, false,
                        msg, invdeg + (long long)r * NNODES,
                        Wl[l] + (long long)r * K * HDIM, nullptr,
                        bufB + (long long)REL_D[r] * NNODES * HDIM, stream);
        }

        // mean over relations + leaky_relu + layernorm -> next layer input
        for (int t = 0; t < NTYPE; ++t)
            act_norm_kernel<<<(NNODES + 7) / 8, 256, 0, stream>>>(
                bufB + (long long)t * NNODES * HDIM,
                bufA + (long long)t * NNODES * HDIM,
                g[l], be[l], 1.0f / (float)CNT[t], NNODES);
    }

    final_kernel<<<(NNODES + 7) / 8, 256, 0, stream>>>(bufA, Wfc, bfc, out, NNODES);
    (void)in_sizes; (void)n_in; (void)out_size; (void)ws_size;
}
